// ScaledFP8Linear_24988119728555
// MI455X (gfx1250) — matrix operand order repaired and checked by reading, not compiled, not run on hardware
//
#include <hip/hip_runtime.h>

// ---------------------------------------------------------------------------
// ScaledFP8Linear on MI455X (gfx1250, wave32, WMMA + async global->LDS)
//   out[M,N] = x[M,K] @ (weight[N,K] * blockscale)ᵀ + bias[N]
//   M=8192, N=6144, K=4096 (derived at launch), scale block = 128x128
// Pass 1: weight*scale -> bf16 (ws), x -> bf16 (ws)   [memory bound]
// Pass 2: bf16 WMMA GEMM, 128x128 tile / WG, LDS double buffered via
//         global_load_async_to_lds_b128 (ASYNCcnt), no VGPR staging.
// ---------------------------------------------------------------------------

typedef __attribute__((ext_vector_type(16))) __bf16 bf16x16;
typedef __attribute__((ext_vector_type(8)))  __bf16 bf16x8;
typedef __attribute__((ext_vector_type(8)))  float  f32x8;

__device__ __forceinline__ unsigned int f32_to_bf16_rne(float f) {
  unsigned int u = __float_as_uint(f);
  u += 0x7FFFu + ((u >> 16) & 1u);
  return u >> 16;
}

__device__ __forceinline__ unsigned int pack2bf(float a, float b) {
  return f32_to_bf16_rne(a) | (f32_to_bf16_rne(b) << 16);
}

// Async global->LDS, GVS addressing: mem = SGPR64 + VGPR_I32 offset.
// vdst VGPR carries the LDS byte address (low 32 bits of flat shared addr).
__device__ __forceinline__ void async_load_b128(unsigned lds, unsigned voff,
                                                unsigned long long sbase) {
  asm volatile("global_load_async_to_lds_b128 %0, %1, %2"
               :: "v"(lds), "v"(voff), "s"(sbase)
               : "memory");
}

__device__ __forceinline__ void wait_async0() {
  asm volatile("s_wait_asynccnt 0x0" ::: "memory");
}

// ---- Pass 1a: wb[o,k] = bf16( weight[o,k] * scale[o>>7, k>>7] ) ------------
__global__ __launch_bounds__(256) void prep_weight_bf16(
    const float* __restrict__ w, const float* __restrict__ scale,
    unsigned short* __restrict__ wb, int K, int SW, int total8) {
  int t = blockIdx.x * 256 + threadIdx.x;
  if (t >= total8) return;
  size_t base = (size_t)t * 8;
  int o = (int)(base / (unsigned)K);
  int i = (int)(base - (size_t)o * (size_t)K);
  float s = scale[(o >> 7) * SW + (i >> 7)];
  const float4* p = (const float4*)(w + base);
  float4 a = p[0];
  float4 b = p[1];
  uint4 r;
  r.x = pack2bf(a.x * s, a.y * s);
  r.y = pack2bf(a.z * s, a.w * s);
  r.z = pack2bf(b.x * s, b.y * s);
  r.w = pack2bf(b.z * s, b.w * s);
  *(uint4*)(wb + base) = r;
}

// ---- Pass 1b: xb = bf16(x) -------------------------------------------------
__global__ __launch_bounds__(256) void prep_x_bf16(
    const float* __restrict__ x, unsigned short* __restrict__ xb, int total8) {
  int t = blockIdx.x * 256 + threadIdx.x;
  if (t >= total8) return;
  size_t base = (size_t)t * 8;
  const float4* p = (const float4*)(x + base);
  float4 a = p[0];
  float4 b = p[1];
  uint4 r;
  r.x = pack2bf(a.x, a.y);
  r.y = pack2bf(a.z, a.w);
  r.z = pack2bf(b.x, b.y);
  r.w = pack2bf(b.z, b.w);
  *(uint4*)(xb + base) = r;
}

// ---- Pass 2: bf16 WMMA GEMM ------------------------------------------------
// Block tile 128(M) x 128(N); 8 waves in 2(M) x 4(N); wave tile 64x32 =
// 4x2 WMMA 16x16 tiles; K staged 64 wide, double-buffered LDS fed by
// async global->LDS.
__global__ __launch_bounds__(256) void gemm_bf16_wmma(
    const unsigned short* __restrict__ xb, const unsigned short* __restrict__ wb,
    const float* __restrict__ bias, float* __restrict__ out,
    int M, int N, int K) {
  __shared__ __align__(32) unsigned short Alds[2][128 * 64];
  __shared__ __align__(32) unsigned short Blds[2][128 * 64];

  const int tid  = threadIdx.x;
  const int lane = tid & 31;
  const int wv   = tid >> 5;
  const int wm   = wv >> 2;   // 0..1  (M direction)
  const int wn   = wv & 3;    // 0..3  (N direction)
  const int l15  = lane & 15;
  const int hi   = lane >> 4; // lane half-select (matches WMMA striping)

  const int m_base = blockIdx.y * 128;
  const int n_base = blockIdx.x * 128;
  const int stages = K >> 6;

  // Per-thread transfer slots: u = tid + 256*i -> (row 0..127, seg 0..7),
  // each slot moves 16 bytes (8 bf16). Stage-invariant offsets:
  unsigned voff[4];   // global byte offset within the 128-row K-stripe
  unsigned loff[4];   // LDS byte offset within one buffer
#pragma unroll
  for (int i = 0; i < 4; ++i) {
    int u = tid + 256 * i;
    int row = u >> 3, seg = u & 7;
    voff[i] = (unsigned)((row * K + seg * 8) * 2);
    loff[i] = (unsigned)((row * 64 + seg * 8) * 2);
  }
  // LDS byte addresses (flat addr low 32 bits == LDS offset, ISA 10.2)
  const unsigned aBase = (unsigned)(uintptr_t)&Alds[0][0];
  const unsigned bBase = (unsigned)(uintptr_t)&Blds[0][0];
  const unsigned bufStride = 128 * 64 * 2;  // 16 KB

  // Uniform global byte bases for this block's tile rows
  const unsigned long long xByte =
      (unsigned long long)(uintptr_t)xb + (unsigned long long)m_base * K * 2ull;
  const unsigned long long wByte =
      (unsigned long long)(uintptr_t)wb + (unsigned long long)n_base * K * 2ull;

  f32x8 acc[4][2];
#pragma unroll
  for (int r = 0; r < 4; ++r)
#pragma unroll
    for (int c = 0; c < 2; ++c) {
      f32x8 z = {0.f, 0.f, 0.f, 0.f, 0.f, 0.f, 0.f, 0.f};
      acc[r][c] = z;
    }

  // prologue: stage 0 -> LDS buf 0 (async), wait own transfers, barrier
#pragma unroll
  for (int i = 0; i < 4; ++i) {
    async_load_b128(aBase + loff[i], voff[i], xByte);
    async_load_b128(bBase + loff[i], voff[i], wByte);
  }
  wait_async0();
  __syncthreads();

  for (int s = 0; s < stages; ++s) {
    const bool more = (s + 1) < stages;
    if (more) {
      // issue stage s+1 into the other buffer; only the SGPR base advances
      const unsigned bsel = ((s + 1) & 1) ? bufStride : 0u;
      const unsigned long long kadd = (unsigned long long)(s + 1) * 128ull; // 64 bf16
      const unsigned long long xs = xByte + kadd;
      const unsigned long long ws = wByte + kadd;
#pragma unroll
      for (int i = 0; i < 4; ++i) {
        async_load_b128(aBase + bsel + loff[i], voff[i], xs);
        async_load_b128(bBase + bsel + loff[i], voff[i], ws);
      }
    }

    const unsigned short* Ab = Alds[s & 1];
    const unsigned short* Bb = Blds[s & 1];
#pragma unroll
    for (int kk = 0; kk < 64; kk += 32) {
      bf16x16 afr[4], bfr[2];
      // A 16x32 fragment: lanes 0-15 rows M=0..15 K[0..7]+K[16..23],
      //                   lanes 16-31 same rows   K[8..15]+K[24..31]
#pragma unroll
      for (int r = 0; r < 4; ++r) {
        int off = (wm * 64 + r * 16 + l15) * 64 + kk + hi * 8;
        union { bf16x16 v; bf16x8 h[2]; } au;
        au.h[0] = *(const bf16x8*)&Ab[off];
        au.h[1] = *(const bf16x8*)&Ab[off + 16];
        afr[r] = au.v;
      }
      // B 32x16 fragment: lanes 0-15 col N, K[0..7]+K[16..23]; lanes 16-31 K[8..15]+K[24..31]
#pragma unroll
      for (int c = 0; c < 2; ++c) {
        int off = (wn * 32 + c * 16 + l15) * 64 + kk + hi * 8;
        union { bf16x16 v; bf16x8 h[2]; } bu;
        bu.h[0] = *(const bf16x8*)&Bb[off];
        bu.h[1] = *(const bf16x8*)&Bb[off + 16];
        bfr[c] = bu.v;
      }
#pragma unroll
      for (int r = 0; r < 4; ++r)
#pragma unroll
        for (int c = 0; c < 2; ++c)
          acc[r][c] = __builtin_amdgcn_wmma_f32_16x16x32_bf16(
              false, afr[r], false, bfr[c], (short)0, acc[r][c], false, false);
    }

    if (more) {
      wait_async0();     // own async transfers into next buffer done
      __syncthreads();   // all waves' transfers visible; prev reads retired
    }
  }

  // epilogue: C/D layout — lane half hi gives M=hi*8+v, col = l15; add bias
#pragma unroll
  for (int r = 0; r < 4; ++r) {
    int m = m_base + wm * 64 + r * 16 + hi * 8;
#pragma unroll
    for (int c = 0; c < 2; ++c) {
      int n = n_base + wn * 32 + c * 16 + l15;
      float bv = bias[n];
#pragma unroll
      for (int v = 0; v < 8; ++v)
        out[(size_t)(m + v) * N + n] = acc[r][c][v] + bv;
    }
  }
}

extern "C" void kernel_launch(void* const* d_in, const int* in_sizes, int n_in,
                              void* d_out, int out_size, void* d_ws, size_t ws_size,
                              hipStream_t stream) {
  const float* x     = (const float*)d_in[0];
  const float* w     = (const float*)d_in[1];
  const float* scale = (const float*)d_in[2];
  const float* bias  = (const float*)d_in[3];
  float* out = (float*)d_out;

  const int N  = in_sizes[3];            // 6144
  const int K  = in_sizes[1] / N;        // 4096
  const int M  = in_sizes[0] / K;        // 8192
  const int SW = K / 128;                // scale cols (128x128 blocks)

  unsigned short* wb = (unsigned short*)d_ws;         // N*K bf16 (48 MB)
  unsigned short* xb = wb + (size_t)N * (size_t)K;    // M*K bf16 (64 MB)

  {
    int total8 = (N * K) / 8;
    prep_weight_bf16<<<(total8 + 255) / 256, 256, 0, stream>>>(w, scale, wb, K, SW, total8);
  }
  {
    int total8 = (M * K) / 8;
    prep_x_bf16<<<(total8 + 255) / 256, 256, 0, stream>>>(x, xb, total8);
  }
  dim3 grid(N / 128, M / 128);  // (48, 64)
  gemm_bf16_wmma<<<grid, 256, 0, stream>>>(xb, wb, bias, out, M, N, K);
}
